// Receiver_7842610283408
// MI455X (gfx1250) — matrix operand order repaired and checked by reading, not compiled, not run on hardware
//
#include <hip/hip_runtime.h>
#include <stdint.h>

#define Hd   1024
#define Ed   512
#define Bsz  512
#define Tlen 64
#define NOb  10
#define FEAT 256
#define G4H  4096
#define EPSf 1e-5f

// GEMM tiling: block = 64M x 256N, 8 waves of 32M x 64N, K step 32
#define BM 64
#define BN 256

typedef __attribute__((ext_vector_type(16))) __bf16 v16bf;
typedef __attribute__((ext_vector_type(8)))  float  v8f;
typedef unsigned int u32x4  __attribute__((ext_vector_type(4)));
typedef unsigned int uint32x4 __attribute__((ext_vector_type(4)));
typedef int          int32x4  __attribute__((ext_vector_type(4)));
typedef int          int32x8  __attribute__((ext_vector_type(8)));

union Frag32B { u32x4 q[2]; v16bf v; };

__device__ __forceinline__ unsigned short f2bf(float f) {
  unsigned int u = __float_as_uint(f);
  u += 0x7FFFu + ((u >> 16) & 1u);          // round-to-nearest-even
  return (unsigned short)(u >> 16);
}
__device__ __forceinline__ float sigm(float x) { return 1.0f / (1.0f + __expf(-x)); }

__device__ __forceinline__ float block_sum(float v, float* sm) {
  int tid = threadIdx.x;
  sm[tid] = v; __syncthreads();
  for (int s = 128; s > 0; s >>= 1) { if (tid < s) sm[tid] += sm[tid + s]; __syncthreads(); }
  float r = sm[0]; __syncthreads();
  return r;
}

#if __has_builtin(__builtin_amdgcn_tensor_load_to_lds)
#define HAVE_TDM 1
#else
#define HAVE_TDM 0
#endif

// ---- TDM 2D tile DMA: global [rows x K] bf16 (row stride = K elems) -> contiguous LDS rows of 64B
__device__ __forceinline__ void tdm_load_2d(unsigned lds_byte_addr, const unsigned short* gptr,
                                            unsigned tensor_d0, unsigned tensor_d1,
                                            unsigned tile_d1, unsigned long long stride0) {
#if HAVE_TDM
  unsigned long long ga = (unsigned long long)(uintptr_t)gptr;
  uint32x4 g0;
  g0[0] = 1u;                                               // count=1, user mode
  g0[1] = lds_byte_addr;                                    // LDS dest (bytes)
  g0[2] = (unsigned)(ga & 0xFFFFFFFFu);                     // global_addr[31:0]
  g0[3] = (unsigned)((ga >> 32) & 0x01FFFFFFu) | (2u << 30);// global_addr[56:32] | type=2
  int32x8 g1;
  g1[0] = (int)(1u << 16);                                  // data_size=1 (2 bytes); mask=0
  g1[1] = (int)((tensor_d0 & 0xFFFFu) << 16);               // tensor_dim0[15:0]
  g1[2] = (int)(((tensor_d0 >> 16) & 0xFFFFu) | ((tensor_d1 & 0xFFFFu) << 16));
  g1[3] = (int)(((tensor_d1 >> 16) & 0xFFFFu) | (32u << 16));   // tile_dim0 = 32 elems
  g1[4] = (int)(tile_d1 & 0xFFFFu);                         // tile_dim1 rows; tile_dim2=0
  g1[5] = (int)(stride0 & 0xFFFFFFFFull);                   // tensor_dim0_stride[31:0]
  g1[6] = (int)((stride0 >> 32) & 0xFFFFull);               // stride[47:32]; dim1_stride=0
  g1[7] = 0;
  int32x4 z4 = {0, 0, 0, 0};
#if __clang_major__ >= 23
  int32x8 z8 = {0, 0, 0, 0, 0, 0, 0, 0};
  __builtin_amdgcn_tensor_load_to_lds(g0, g1, z4, z4, z8, 0);
#else
  __builtin_amdgcn_tensor_load_to_lds(g0, g1, z4, z4, 0);
#endif
#else
  // Fallback: vector copy, lane per row (caller passes per-wave slices of <=32 rows)
  int lane = threadIdx.x & 31;
  if (lane < (int)tile_d1) {
    const u32x4* src = (const u32x4*)(gptr + (size_t)lane * (size_t)stride0);
    u32x4* dst = (u32x4*)(uintptr_t)(lds_byte_addr + lane * 64);
    dst[0] = src[0]; dst[1] = src[1]; dst[2] = src[2]; dst[3] = src[3];
  }
#endif
}

// ---------------- weight fp32 -> bf16 ----------------
__global__ void cvt_bf16(const float* __restrict__ src, unsigned short* __restrict__ dst, int n) {
  int i = blockIdx.x * blockDim.x + threadIdx.x;
  int stride = gridDim.x * blockDim.x;
  for (; i < n; i += stride) dst[i] = f2bf(src[i]);
}

// ---------------- zero LSTM state ----------------
__global__ void init_state(float* c0, float* c1, unsigned short* h0b, unsigned short* h1b) {
  int i = blockIdx.x * blockDim.x + threadIdx.x;
  if (i < Bsz * Hd) { c0[i] = 0.f; c1[i] = 0.f; h0b[i] = 0; h1b[i] = 0; }
}

// ---------------- embedding gather for step t ----------------
__global__ void embed_step(const int* __restrict__ msg, const float* __restrict__ emb,
                           unsigned short* __restrict__ xbf, int t) {
  int b = blockIdx.x;
  int tok = msg[b * Tlen + t];
  const float* row = emb + (size_t)tok * Ed;
  for (int e = threadIdx.x; e < Ed; e += blockDim.x) xbf[b * Ed + e] = f2bf(row[e]);
}

// ---------------- dual bf16 WMMA GEMM: C = X @ W^T  (M=512, N=4096) ----------------
// TDM DMA stages K-tiles (A: 64x32, B: 256x32 bf16) into double-buffered LDS; WMMA reads LDS.
__global__ void __launch_bounds__(256)
gemm_dual(const unsigned short* __restrict__ X0, const unsigned short* __restrict__ W0,
          float* __restrict__ C0, int K0,
          const unsigned short* __restrict__ X1, const unsigned short* __restrict__ W1,
          float* __restrict__ C1, int K1) {
  __shared__ __align__(16) unsigned char ldsA[2][BM * 64];   //  4 KB x2: 64 rows x 64B
  __shared__ __align__(16) unsigned char ldsB[2][BN * 64];   // 16 KB x2: 256 rows x 64B

  const int lane = threadIdx.x & 31;
  const int wave = threadIdx.x >> 5;   // 0..7
  const int wm   = wave >> 2;          // 0..1 : 32-row half of the 64-row block tile
  const int wn   = wave & 3;           // 0..3 : 64-col quarter of the 256-col block tile
  const int half = lane >> 4;
  const int lr   = lane & 15;
  const int mBase = blockIdx.y * BM;
  const int nBase = blockIdx.x * BN;

  for (int part = 0; part < 2; ++part) {
    const unsigned short* X = part ? X1 : X0;
    const unsigned short* W = part ? W1 : W0;
    float* C    = part ? C1 : C0;
    const int K = part ? K1 : K0;
    const int nk = K >> 5;

    // Each wave DMAs a uniform slice: 32 B-rows + 8 A-rows per K-tile (2 TDM ops/wave/tile).
    const unsigned short* wSlice = W + (size_t)(nBase + wave * 32) * K;
    const unsigned short* xSlice = X + (size_t)(mBase + wave * 8) * K;

    // prologue: tile 0 -> buffer 0
    tdm_load_2d((unsigned)(uintptr_t)&ldsB[0][wave * 32 * 64], wSlice, (unsigned)K, 4096u, 32u,
                (unsigned long long)K);
    tdm_load_2d((unsigned)(uintptr_t)&ldsA[0][wave * 8 * 64], xSlice, (unsigned)K, 512u, 8u,
                (unsigned long long)K);

    v8f acc[2][4] = {};
    for (int kt = 0; kt < nk; ++kt) {
      const int cur = kt & 1;
      if (kt + 1 < nk) {
        const int kk = (kt + 1) * 32;
        tdm_load_2d((unsigned)(uintptr_t)&ldsB[cur ^ 1][wave * 32 * 64], wSlice + kk,
                    (unsigned)K, 4096u, 32u, (unsigned long long)K);
        tdm_load_2d((unsigned)(uintptr_t)&ldsA[cur ^ 1][wave * 8 * 64], xSlice + kk,
                    (unsigned)K, 512u, 8u, (unsigned long long)K);
#if HAVE_TDM
        __builtin_amdgcn_s_wait_tensorcnt(2);   // tile kt done; kt+1 still in flight
      } else {
        __builtin_amdgcn_s_wait_tensorcnt(0);
#endif
      }
      __syncthreads();                          // tile kt visible to all waves

      const unsigned char* bufA = ldsA[cur];
      const unsigned char* bufB = ldsB[cur];
      Frag32B a0, a1, b0, b1, b2, b3;
      const int ar0 = wm * 32 + lr, ar1 = wm * 32 + 16 + lr;
      a0.q[0] = *(const u32x4*)(bufA + ar0 * 64 + half * 32);
      a0.q[1] = *(const u32x4*)(bufA + ar0 * 64 + half * 32 + 16);
      a1.q[0] = *(const u32x4*)(bufA + ar1 * 64 + half * 32);
      a1.q[1] = *(const u32x4*)(bufA + ar1 * 64 + half * 32 + 16);
      const int br = wn * 64 + lr;
      b0.q[0] = *(const u32x4*)(bufB + (br +  0) * 64 + half * 32);
      b0.q[1] = *(const u32x4*)(bufB + (br +  0) * 64 + half * 32 + 16);
      b1.q[0] = *(const u32x4*)(bufB + (br + 16) * 64 + half * 32);
      b1.q[1] = *(const u32x4*)(bufB + (br + 16) * 64 + half * 32 + 16);
      b2.q[0] = *(const u32x4*)(bufB + (br + 32) * 64 + half * 32);
      b2.q[1] = *(const u32x4*)(bufB + (br + 32) * 64 + half * 32 + 16);
      b3.q[0] = *(const u32x4*)(bufB + (br + 48) * 64 + half * 32);
      b3.q[1] = *(const u32x4*)(bufB + (br + 48) * 64 + half * 32 + 16);

      acc[0][0] = __builtin_amdgcn_wmma_f32_16x16x32_bf16(false, a0.v, false, b0.v, (short)0, acc[0][0], false, false);
      acc[0][1] = __builtin_amdgcn_wmma_f32_16x16x32_bf16(false, a0.v, false, b1.v, (short)0, acc[0][1], false, false);
      acc[0][2] = __builtin_amdgcn_wmma_f32_16x16x32_bf16(false, a0.v, false, b2.v, (short)0, acc[0][2], false, false);
      acc[0][3] = __builtin_amdgcn_wmma_f32_16x16x32_bf16(false, a0.v, false, b3.v, (short)0, acc[0][3], false, false);
      acc[1][0] = __builtin_amdgcn_wmma_f32_16x16x32_bf16(false, a1.v, false, b0.v, (short)0, acc[1][0], false, false);
      acc[1][1] = __builtin_amdgcn_wmma_f32_16x16x32_bf16(false, a1.v, false, b1.v, (short)0, acc[1][1], false, false);
      acc[1][2] = __builtin_amdgcn_wmma_f32_16x16x32_bf16(false, a1.v, false, b2.v, (short)0, acc[1][2], false, false);
      acc[1][3] = __builtin_amdgcn_wmma_f32_16x16x32_bf16(false, a1.v, false, b3.v, (short)0, acc[1][3], false, false);

      __syncthreads();                          // all waves done with buf[cur] before reuse
    }

    // C layout: VGPR r -> row (+half*8+r), lane lr -> col
    #pragma unroll
    for (int i = 0; i < 2; ++i) {
      #pragma unroll
      for (int r = 0; r < 8; ++r) {
        size_t row = (size_t)(mBase + wm * 32 + i * 16 + half * 8 + r) * G4H;
        const int col = nBase + wn * 64 + lr;
        C[row + col +  0] = acc[i][0][r];
        C[row + col + 16] = acc[i][1][r];
        C[row + col + 32] = acc[i][2][r];
        C[row + col + 48] = acc[i][3][r];
      }
    }
  }
}

// ------------- fused LN(gates) + LSTM cell + LN(cell) + bf16(h) + encoded gather -------------
__global__ void __launch_bounds__(256)
lstm_pointwise(const float* __restrict__ A, const float* __restrict__ Bm,
               const float* __restrict__ gi, const float* __restrict__ bi,
               const float* __restrict__ gh, const float* __restrict__ bh,
               const float* __restrict__ gc, const float* __restrict__ bc,
               float* __restrict__ c, unsigned short* __restrict__ hbf,
               float* __restrict__ encoded, const int* __restrict__ lengths, int t) {
  __shared__ float sm[256];
  const int b = blockIdx.x, tid = threadIdx.x;
  const float* Ar = A  + (size_t)b * G4H;
  const float* Br = Bm + (size_t)b * G4H;

  float sA = 0, sAA = 0, sB = 0, sBB = 0;
  for (int i = tid; i < G4H; i += 256) {
    float a = Ar[i], bb = Br[i];
    sA += a; sAA += a * a; sB += bb; sBB += bb * bb;
  }
  sA  = block_sum(sA, sm);  sAA = block_sum(sAA, sm);
  sB  = block_sum(sB, sm);  sBB = block_sum(sBB, sm);
  const float muA = sA * (1.f / G4H), muB = sB * (1.f / G4H);
  const float rA = rsqrtf(sAA * (1.f / G4H) - muA * muA + EPSf);
  const float rB = rsqrtf(sBB * (1.f / G4H) - muB * muB + EPSf);

  float cand[4], og[4];
  float sc = 0, scc = 0;
  #pragma unroll
  for (int u = 0; u < 4; ++u) {
    int j = tid + u * 256;
    float vi = (Ar[0*Hd+j]-muA)*rA*gi[0*Hd+j]+bi[0*Hd+j] + (Br[0*Hd+j]-muB)*rB*gh[0*Hd+j]+bh[0*Hd+j];
    float vf = (Ar[1*Hd+j]-muA)*rA*gi[1*Hd+j]+bi[1*Hd+j] + (Br[1*Hd+j]-muB)*rB*gh[1*Hd+j]+bh[1*Hd+j];
    float vg = (Ar[2*Hd+j]-muA)*rA*gi[2*Hd+j]+bi[2*Hd+j] + (Br[2*Hd+j]-muB)*rB*gh[2*Hd+j]+bh[2*Hd+j];
    float vo = (Ar[3*Hd+j]-muA)*rA*gi[3*Hd+j]+bi[3*Hd+j] + (Br[3*Hd+j]-muB)*rB*gh[3*Hd+j]+bh[3*Hd+j];
    float cd = sigm(vf) * c[(size_t)b*Hd + j] + sigm(vi) * tanhf(vg);
    cand[u] = cd; og[u] = vo;
    sc += cd; scc += cd * cd;
  }
  sc = block_sum(sc, sm); scc = block_sum(scc, sm);
  const float muC = sc * (1.f / Hd);
  const float rC  = rsqrtf(scc * (1.f / Hd) - muC * muC + EPSf);
  const bool sel = (encoded != nullptr) && (lengths[b] - 1 == t);
  #pragma unroll
  for (int u = 0; u < 4; ++u) {
    int j = tid + u * 256;
    float cy = (cand[u] - muC) * rC * gc[j] + bc[j];
    float hy = sigm(og[u]) * tanhf(cy);
    c[(size_t)b*Hd + j]   = cy;
    hbf[(size_t)b*Hd + j] = f2bf(hy);
    if (sel) encoded[(size_t)b*Hd + j] = hy;
  }
}

// ---------------- head: fc1/tanh, prod, einsum, out_w head, softmax factor, sa ----------------
__global__ void __launch_bounds__(256)
head_kernel(const float* __restrict__ inp, const float* __restrict__ enc,
            const float* __restrict__ fc1_w, const float* __restrict__ fc1_b,
            const float* __restrict__ out_w, const float* __restrict__ out_b,
            const float* __restrict__ sa_w, const float* __restrict__ sa_b,
            float* __restrict__ out0, float* __restrict__ out1) {
  __shared__ float sm[256];
  __shared__ float dots[NOb + 1];
  __shared__ float sa[2];
  const int b = blockIdx.x, tid = threadIdx.x;
  const float* er = enc + (size_t)b * Hd;
  float prod[4], encv[4];
  #pragma unroll
  for (int u = 0; u < 4; ++u) { prod[u] = 1.f; encv[u] = er[tid + u * 256]; }

  for (int o = 0; o < NOb; ++o) {
    const float* irow = inp + ((size_t)b * NOb + o) * FEAT;
    float part = 0;
    #pragma unroll
    for (int u = 0; u < 4; ++u) {
      int hh = tid + u * 256;
      const float* wrow = fc1_w + (size_t)hh * FEAT;
      float acc = fc1_b[hh];
      #pragma unroll 8
      for (int k = 0; k < FEAT; ++k) acc += irow[k] * wrow[k];
      float ei = tanhf(acc);
      prod[u] *= ei;
      part += ei * encv[u];
    }
    float d = block_sum(part, sm);
    if (tid == 0) dots[o] = d;
  }
  float part = 0;
  #pragma unroll
  for (int u = 0; u < 4; ++u) part += prod[u] * encv[u];
  float d = block_sum(part, sm);
  if (tid == 0) dots[NOb] = d;

  float p0 = 0, p1 = 0;
  #pragma unroll
  for (int u = 0; u < 4; ++u) { int hh = tid + u * 256; p0 += encv[u] * sa_w[hh]; p1 += encv[u] * sa_w[Hd + hh]; }
  p0 = block_sum(p0, sm); p1 = block_sum(p1, sm);
  if (tid == 0) { sa[0] = p0 + sa_b[0]; sa[1] = p1 + sa_b[1]; }
  __syncthreads();

  if (tid < 2) out1[b * 2 + tid] = sa[tid];
  if (tid < NOb + 2) {
    float acc = out_b[tid];
    #pragma unroll
    for (int j = 0; j < NOb + 1; ++j) acc += out_w[tid * (NOb + 1) + j] * dots[j];
    float m = fmaxf(sa[0], sa[1]);
    float e0 = __expf(sa[0] - m), e1 = __expf(sa[1] - m);
    float f = (tid < NOb) ? e0 / (e0 + e1) : e1 / (e0 + e1);
    out0[b * (NOb + 2) + tid] = acc * f;
  }
}

extern "C" void kernel_launch(void* const* d_in, const int* in_sizes, int n_in,
                              void* d_out, int out_size, void* d_ws, size_t ws_size,
                              hipStream_t stream) {
  const int*   message = (const int*)  d_in[0];
  const float* inp     = (const float*)d_in[1];
  const int*   lengths = (const int*)  d_in[2];
  const float* emb     = (const float*)d_in[3];
  const float* wih0    = (const float*)d_in[4];
  const float* whh0    = (const float*)d_in[5];
  const float* wih1    = (const float*)d_in[6];
  const float* whh1    = (const float*)d_in[7];
  const float* gi0 = (const float*)d_in[8],  *bi0 = (const float*)d_in[9];
  const float* gh0 = (const float*)d_in[10], *bh0 = (const float*)d_in[11];
  const float* gc0 = (const float*)d_in[12], *bc0 = (const float*)d_in[13];
  const float* gi1 = (const float*)d_in[14], *bi1 = (const float*)d_in[15];
  const float* gh1 = (const float*)d_in[16], *bh1 = (const float*)d_in[17];
  const float* gc1 = (const float*)d_in[18], *bc1 = (const float*)d_in[19];
  const float* fc1_w = (const float*)d_in[20], *fc1_b = (const float*)d_in[21];
  const float* out_w = (const float*)d_in[22], *out_b = (const float*)d_in[23];
  const float* sa_w  = (const float*)d_in[24], *sa_b  = (const float*)d_in[25];

  char* ws = (char*)d_ws;
  size_t off = 0;
  auto alloc = [&](size_t bytes) -> void* {
    void* p = ws + off; off += (bytes + 255) & ~(size_t)255; return p;
  };
  unsigned short* wih0b = (unsigned short*)alloc((size_t)G4H * Ed * 2);
  unsigned short* whh0b = (unsigned short*)alloc((size_t)G4H * Hd * 2);
  unsigned short* wih1b = (unsigned short*)alloc((size_t)G4H * Hd * 2);
  unsigned short* whh1b = (unsigned short*)alloc((size_t)G4H * Hd * 2);
  unsigned short* xbf   = (unsigned short*)alloc((size_t)Bsz * Ed * 2);
  unsigned short* h0b   = (unsigned short*)alloc((size_t)Bsz * Hd * 2);
  unsigned short* h1b   = (unsigned short*)alloc((size_t)Bsz * Hd * 2);
  float* c0   = (float*)alloc((size_t)Bsz * Hd * 4);
  float* c1   = (float*)alloc((size_t)Bsz * Hd * 4);
  float* Abuf = (float*)alloc((size_t)Bsz * G4H * 4);
  float* Bbuf = (float*)alloc((size_t)Bsz * G4H * 4);
  float* encoded = (float*)alloc((size_t)Bsz * Hd * 4);
  (void)ws_size; (void)in_sizes; (void)n_in; (void)out_size;

  cvt_bf16<<<2048, 256, 0, stream>>>(wih0, wih0b, G4H * Ed);
  cvt_bf16<<<4096, 256, 0, stream>>>(whh0, whh0b, G4H * Hd);
  cvt_bf16<<<4096, 256, 0, stream>>>(wih1, wih1b, G4H * Hd);
  cvt_bf16<<<4096, 256, 0, stream>>>(whh1, whh1b, G4H * Hd);
  init_state<<<(Bsz * Hd) / 256, 256, 0, stream>>>(c0, c1, h0b, h1b);

  dim3 ggrid(G4H / BN, Bsz / BM);  // (16, 8)
  for (int t = 0; t < Tlen; ++t) {
    embed_step<<<Bsz, 256, 0, stream>>>(message, emb, xbf, t);
    gemm_dual<<<ggrid, 256, 0, stream>>>(xbf, wih0b, Abuf, Ed, h0b, whh0b, Bbuf, Hd);
    lstm_pointwise<<<Bsz, 256, 0, stream>>>(Abuf, Bbuf, gi0, bi0, gh0, bh0, gc0, bc0,
                                            c0, h0b, nullptr, lengths, t);
    gemm_dual<<<ggrid, 256, 0, stream>>>(h0b, wih1b, Abuf, Hd, h1b, whh1b, Bbuf, Hd);
    lstm_pointwise<<<Bsz, 256, 0, stream>>>(Abuf, Bbuf, gi1, bi1, gh1, bh1, gc1, bc1,
                                            c1, h1b, encoded, lengths, t);
  }
  head_kernel<<<Bsz, 256, 0, stream>>>(inp, encoded, fc1_w, fc1_b, out_w, out_b,
                                       sa_w, sa_b, (float*)d_out, (float*)d_out + Bsz * (NOb + 2));
}
